// SlotAttention_44787918962821
// MI455X (gfx1250) — matrix operand order repaired and checked by reading, not compiled, not run on hardware
//
#include <hip/hip_runtime.h>
#include <hip/hip_bf16.h>

// ---------------------------------------------------------------------------
// Slot Attention for MI455X (gfx1250, wave32, WMMA).
// Heavy GEMMs (k/v projection, q.k^T logits, attn.v updates) use
// v_wmma_f32_16x16x32_f16 (f16 inputs, f32 accumulate). k stored [B,N,64] f16,
// v stored transposed [B,64,N] f16 so WMMA B-fragments are contiguous lane
// loads. Slot-axis softmax is done per-lane directly on the WMMA C fragment.
// Per-wave LDS staging is ordered with wave-local s_wait_dscnt (DS ops from a
// wave are in-order), avoiding block barriers in hot loops. Next-tile k/v are
// prefetched with global_prefetch.
// ---------------------------------------------------------------------------

typedef __attribute__((ext_vector_type(16))) _Float16 v16h;
typedef __attribute__((ext_vector_type(8)))  _Float16 v8h;
typedef __attribute__((ext_vector_type(8)))  float    v8f;
typedef __attribute__((ext_vector_type(4)))  float    v4f;

#define B_   128
#define N_   4096
#define S_   7
#define D_   64
#define H_   128
#define EPS_ATTN 1e-8f

// Wave-local LDS ordering: DS ops from one wave execute in-order in the LDS
// pipeline; waiting DScnt==0 makes prior ds_stores visible to this wave's
// subsequent ds_loads. "memory" clobber stops compiler reordering.
#define WAVE_LDS_FENCE() asm volatile("s_wait_dscnt 0x0" ::: "memory")
#define COMPILER_FENCE() asm volatile("" ::: "memory")

__device__ __forceinline__ v8f wmma_f16(v16h a, v16h b, v8f c) {
  // (neg_a, A, neg_b, B, c_mod, C, reuse_a, reuse_b)
  return __builtin_amdgcn_wmma_f32_16x16x32_f16(false, a, false, b, (short)0, c,
                                                false, false);
}

// A-fragment (16x32 f16, M=lane&15, kh=lane>>4): K = kbase+kh*16..+15,
// two 16B loads from a row-contiguous buffer.
__device__ __forceinline__ v16h make_a_frag(const _Float16* row_base, int kh,
                                            int kbase) {
  v8h lo = *(const v8h*)(row_base + kbase + kh * 16);
  v8h hi = *(const v8h*)(row_base + kbase + kh * 16 + 8);
  v16h a;
#pragma unroll
  for (int i = 0; i < 8; ++i) { a[i] = lo[i]; a[8 + i] = hi[i]; }
  return a;
}

// ---------------------------------------------------------------------------
// slots = mu + exp(log_sigma) * noise   (written into d_out, updated in place)
// ---------------------------------------------------------------------------
__global__ void k_init_slots(const float* __restrict__ noise,
                             const float* __restrict__ mu,
                             const float* __restrict__ lsig,
                             float* __restrict__ slots, int total) {
  int i = blockIdx.x * blockDim.x + threadIdx.x;
  if (i < total) {
    int d = i & (D_ - 1);
    slots[i] = mu[d] + __expf(lsig[d]) * noise[i];
  }
}

// ---------------------------------------------------------------------------
// Wk, Wv -> f16 (row-major [64][64]); B-fragment loads read W[d][c..] directly.
// ---------------------------------------------------------------------------
__global__ void k_w16(const float* __restrict__ wk, const float* __restrict__ wv,
                      _Float16* __restrict__ wk16, _Float16* __restrict__ wv16) {
  int i = blockIdx.x * blockDim.x + threadIdx.x;
  if (i < D_ * D_) {
    wk16[i] = (_Float16)wk[i];
    wv16[i] = (_Float16)wv[i];
  }
}

// ---------------------------------------------------------------------------
// Fused LayerNorm(inputs) + k/v projection (WMMA).
// One wave = one 16-row tile. Each lane holds 32 of its row's 64 values
// (exactly the A-fragment K positions); row stats via one shfl_xor(16).
// C fragments are staged through a per-wave LDS tile so global writes are
// b128 bursts:  k -> [B*N][64] f16,  v -> transposed [B][64][N] f16.
// ---------------------------------------------------------------------------
__global__ void k_project(const float* __restrict__ x,
                          const float* __restrict__ g,
                          const float* __restrict__ bta,
                          const _Float16* __restrict__ wk,
                          const _Float16* __restrict__ wv,
                          _Float16* __restrict__ kf,
                          _Float16* __restrict__ vT) {
  __shared__ __align__(32) _Float16 stage[8][1024];  // 2KB per wave
  const int wave = threadIdx.x >> 5, lane = threadIdx.x & 31;
  const int rowbase = (blockIdx.x * 8 + wave) * 16;
  const int b = rowbase / N_, nloc = rowbase % N_;
  const int M = lane & 15, kh = lane >> 4;
  _Float16* st = &stage[wave][0];

  // ---- fused LayerNorm -> f16 A fragments ----
  const float* xrow = x + (size_t)(rowbase + M) * D_;
  v8f xv[4], gv[4], bv[4];
  float s = 0.f, ss = 0.f;
#pragma unroll
  for (int c2 = 0; c2 < 4; ++c2) {
    const int o = (c2 >> 1) * 32 + kh * 16 + (c2 & 1) * 8;  // K offsets of A fragments a0,a1
    xv[c2] = *(const v8f*)(xrow + o);
    gv[c2] = *(const v8f*)(g + o);
    bv[c2] = *(const v8f*)(bta + o);
#pragma unroll
    for (int i = 0; i < 8; ++i) { s += xv[c2][i]; ss += xv[c2][i] * xv[c2][i]; }
  }
  // lanes l and l^16 together hold the full 64-value row
  s += __shfl_xor(s, 16, 32);
  ss += __shfl_xor(ss, 16, 32);
  const float mean = s * (1.f / 64.f);
  const float var = ss * (1.f / 64.f) - mean * mean;
  const float rstd = rsqrtf(var + 1e-5f);

  v16h a0, a1;
#pragma unroll
  for (int c2 = 0; c2 < 4; ++c2)
#pragma unroll
    for (int i = 0; i < 8; ++i) {
      _Float16 hv = (_Float16)((xv[c2][i] - mean) * rstd * gv[c2][i] + bv[c2][i]);
      if (c2 < 2) a0[c2 * 8 + i] = hv; else a1[(c2 - 2) * 8 + i] = hv;
    }

  // ---- k = x_ln @ Wk^T : stage [n16][d64] tile in LDS, burst out ----
#pragma unroll
  for (int dt = 0; dt < 4; ++dt) {
    v8f c = {};
    c = wmma_f16(a0, *(const v16h*)(wk + (size_t)(dt * 16 + M) * D_ + kh * 16), c);
    c = wmma_f16(a1, *(const v16h*)(wk + (size_t)(dt * 16 + M) * D_ + 32 + kh * 16), c);
#pragma unroll
    for (int r = 0; r < 8; ++r)
      st[(r + 8 * kh) * 64 + dt * 16 + M] = (_Float16)c[r];  // [n][d]
  }
  WAVE_LDS_FENCE();
  {
    // lane -> row (lane&15), 32-half chunk (lane>>4)
    v16h t0 = *(const v16h*)(st + M * 64 + kh * 32);
    v16h t1 = *(const v16h*)(st + M * 64 + kh * 32 + 16);
    *(v16h*)(kf + (size_t)(rowbase + M) * D_ + kh * 32) = t0;
    *(v16h*)(kf + (size_t)(rowbase + M) * D_ + kh * 32 + 16) = t1;
  }
  COMPILER_FENCE();

  // ---- v = x_ln @ Wv^T : stage transposed [d64][n16] tile, burst out ----
#pragma unroll
  for (int dt = 0; dt < 4; ++dt) {
    v8f c = {};
    c = wmma_f16(a0, *(const v16h*)(wv + (size_t)(dt * 16 + M) * D_ + kh * 16), c);
    c = wmma_f16(a1, *(const v16h*)(wv + (size_t)(dt * 16 + M) * D_ + 32 + kh * 16), c);
#pragma unroll
    for (int r = 0; r < 8; ++r)
      st[(dt * 16 + M) * 16 + r + 8 * kh] = (_Float16)c[r];  // [d][n]
  }
  WAVE_LDS_FENCE();
#pragma unroll
  for (int j = 0; j < 2; ++j) {
    const int d = lane * 2 + j;
    v16h t = *(const v16h*)(st + d * 16);
    *(v16h*)(vT + ((size_t)b * D_ + d) * N_ + nloc) = t;
  }
}

// ---------------------------------------------------------------------------
// q = LN(slots) @ Wq^T * D^-0.5, f16, padded to 16 rows (rows 7..15 zero).
// One block per batch; wave s handles slot s.
// ---------------------------------------------------------------------------
__global__ void k_qproj(const float* __restrict__ slots,
                        const float* __restrict__ gs,
                        const float* __restrict__ bs,
                        const float* __restrict__ Wq,
                        _Float16* __restrict__ q16) {
  const int b = blockIdx.x;
  const int wave = threadIdx.x >> 5, lane = threadIdx.x & 31;
  __shared__ __align__(16) float sln[S_][D_];
  if (wave < S_) {
    const float2 xv = *(const float2*)(slots + ((size_t)b * S_ + wave) * D_ + lane * 2);
    float s = xv.x + xv.y, ss = xv.x * xv.x + xv.y * xv.y;
#pragma unroll
    for (int m = 16; m >= 1; m >>= 1) {
      s += __shfl_xor(s, m, 32);
      ss += __shfl_xor(ss, m, 32);
    }
    const float mean = s * (1.f / 64.f);
    const float rstd = rsqrtf(ss * (1.f / 64.f) - mean * mean + 1e-5f);
    const float2 gv = *(const float2*)(gs + lane * 2);
    const float2 bv = *(const float2*)(bs + lane * 2);
    sln[wave][lane * 2] = (xv.x - mean) * rstd * gv.x + bv.x;
    sln[wave][lane * 2 + 1] = (xv.y - mean) * rstd * gv.y + bv.y;
  }
  __syncthreads();
  if (wave < S_) {
#pragma unroll
    for (int j = 0; j < 2; ++j) {
      const int d = lane * 2 + j;
      const float* wrow = Wq + (size_t)d * D_;
      float acc = 0.f;
      for (int c = 0; c < D_; ++c) acc += wrow[c] * sln[wave][c];
      q16[((size_t)b * 16 + wave) * D_ + d] = (_Float16)(acc * 0.125f);  // 64^-0.5
    }
  } else if (wave == 7) {
    for (int idx = lane; idx < 9 * D_; idx += 32)
      q16[((size_t)b * 16 + 7) * D_ + idx] = (_Float16)0.f;
  }
}

// ---------------------------------------------------------------------------
// Attention core. One block per batch, 8 waves; wave w owns n in [512w,512w+512).
//   logits tile (16 slots x 16 cols) : 2x WMMA over K=64
//   softmax over slots               : per-lane on C fragment VGPRs 0..6
//   P -> LDS -> A-fragment           : free transpose, wave-local dscnt fence
//   U += P @ V (4 d-tiles)           : WMMA, vT contiguous B loads
// Normalization sum_n(attn) hoisted: updates = U / rowsum. Cross-wave U and
// rowsum reduced in LDS (ds_add_f32 + block reduction). Next tile's k/v rows
// are prefetched (one row per lane -> whole tile per instruction).
// ---------------------------------------------------------------------------
__global__ void k_attention(const _Float16* __restrict__ q16,
                            const _Float16* __restrict__ kf,
                            const _Float16* __restrict__ vT,
                            float* __restrict__ upd) {
  const int b = blockIdx.x;
  const int wave = threadIdx.x >> 5, lane = threadIdx.x & 31;
  const int M = lane & 15, kh = lane >> 4;

  __shared__ __align__(32) _Float16 pstage[8][16][32];  // per-wave attn tile
  __shared__ __align__(16) float uacc[8][16][64];       // per-wave partial U
  __shared__ float rsum[16];
  if (threadIdx.x < 16) rsum[threadIdx.x] = 0.f;
  __syncthreads();

  const _Float16* qrow = q16 + ((size_t)b * 16 + M) * D_;
  const v16h a0 = make_a_frag(qrow, kh, 0);
  const v16h a1 = make_a_frag(qrow, kh, 32);
  const _Float16* kb = kf + (size_t)b * N_ * D_;
  const _Float16* vb = vT + (size_t)b * D_ * N_;

  v8f U0 = {}, U1 = {}, U2 = {}, U3 = {};
  float rs[7];
#pragma unroll
  for (int r = 0; r < 7; ++r) rs[r] = 0.f;

  const int nw = wave * 512;
  for (int t = 0; t < 16; ++t) {
    const int nb = nw + t * 32;
    if (t + 1 < 16) {
      // one 128B k-row / 128B vT-row per lane covers the whole next tile
      const int np = nb + 32;
      __builtin_prefetch(kb + (size_t)(np + lane) * D_, 0, 0);
      __builtin_prefetch(vb + (size_t)lane * N_ + np, 0, 0);
      __builtin_prefetch(vb + (size_t)(32 + lane) * N_ + np, 0, 0);
    }
#pragma unroll
    for (int half = 0; half < 2; ++half) {
      const int ncol = nb + half * 16 + M;
      v8f c = {};
      c = wmma_f16(a0, *(const v16h*)(kb + (size_t)ncol * D_ + kh * 16), c);
      c = wmma_f16(a1, *(const v16h*)(kb + (size_t)ncol * D_ + 32 + kh * 16), c);
      // softmax over the 7 slots (rows 0..6 live in this lane's VGPRs, kh==0)
      float mx = c[0];
#pragma unroll
      for (int r = 1; r < 7; ++r) mx = fmaxf(mx, c[r]);
      float e[8], sum = 0.f;
#pragma unroll
      for (int r = 0; r < 7; ++r) { e[r] = __expf(c[r] - mx); sum += e[r]; }
      const float inv = 1.f / sum;
#pragma unroll
      for (int r = 0; r < 7; ++r) { e[r] = e[r] * inv + EPS_ATTN; rs[r] += e[r]; }
      e[7] = 0.f;
#pragma unroll
      for (int r = 0; r < 8; ++r)
        pstage[wave][r + 8 * kh][half * 16 + M] = (_Float16)e[r];
    }
    WAVE_LDS_FENCE();  // per-wave pstage write -> read; no block barrier needed
    const v16h pa = make_a_frag(&pstage[wave][M][0], kh, 0);
    U0 = wmma_f16(pa, *(const v16h*)(vb + (size_t)(0 * 16 + M) * N_ + nb + kh * 16), U0);
    U1 = wmma_f16(pa, *(const v16h*)(vb + (size_t)(1 * 16 + M) * N_ + nb + kh * 16), U1);
    U2 = wmma_f16(pa, *(const v16h*)(vb + (size_t)(2 * 16 + M) * N_ + nb + kh * 16), U2);
    U3 = wmma_f16(pa, *(const v16h*)(vb + (size_t)(3 * 16 + M) * N_ + nb + kh * 16), U3);
    COMPILER_FENCE();  // keep next iteration's pstage stores after these reads
  }

#pragma unroll
  for (int r = 0; r < 8; ++r) {
    uacc[wave][r + 8 * kh][0 * 16 + M] = U0[r];
    uacc[wave][r + 8 * kh][1 * 16 + M] = U1[r];
    uacc[wave][r + 8 * kh][2 * 16 + M] = U2[r];
    uacc[wave][r + 8 * kh][3 * 16 + M] = U3[r];
  }
  if (kh == 0) {
#pragma unroll
    for (int r = 0; r < 7; ++r) atomicAdd(&rsum[r], rs[r]);
  }
  __syncthreads();

  // cross-wave reduction: 256 threads x 4 consecutive f32 of the 16x64 tile
  const int idx = threadIdx.x * 4;
  const int row = idx >> 6, col = idx & 63;
  if (row < S_) {
    v4f acc = {};
#pragma unroll
    for (int w = 0; w < 8; ++w) acc += *(const v4f*)&uacc[w][row][col];
    acc *= (1.f / rsum[row]);
    *(v4f*)(upd + ((size_t)b * S_ + row) * D_ + col) = acc;
  }
}

// ---------------------------------------------------------------------------
// GRU cell + LN + MLP residual per (b,s) row. Tiny: plain VALU, 64 threads.
// ---------------------------------------------------------------------------
__global__ void k_gru_mlp(const float* __restrict__ upd, float* __restrict__ slots,
                          const float* __restrict__ W_ih, const float* __restrict__ W_hh,
                          const float* __restrict__ b_ih, const float* __restrict__ b_hh,
                          const float* __restrict__ gm, const float* __restrict__ bm,
                          const float* __restrict__ W1, const float* __restrict__ b1,
                          const float* __restrict__ W2, const float* __restrict__ b2) {
  const int row = blockIdx.x;   // 0 .. B*S-1
  const int j = threadIdx.x;    // 0 .. 63
  __shared__ float u[D_], h[D_], mv[D_], hid[H_];
  u[j] = upd[(size_t)row * D_ + j];
  h[j] = slots[(size_t)row * D_ + j];
  __syncthreads();

  float gi[3], gh[3];
#pragma unroll
  for (int g = 0; g < 3; ++g) {
    const float* wi = W_ih + (size_t)(g * D_ + j) * D_;
    const float* wh = W_hh + (size_t)(g * D_ + j) * D_;
    float ai = b_ih[g * D_ + j], ah = b_hh[g * D_ + j];
    for (int c = 0; c < D_; ++c) { ai += wi[c] * u[c]; ah += wh[c] * h[c]; }
    gi[g] = ai; gh[g] = ah;
  }
  const float r = 1.f / (1.f + __expf(-(gi[0] + gh[0])));
  const float z = 1.f / (1.f + __expf(-(gi[1] + gh[1])));
  const float n = tanhf(gi[2] + r * gh[2]);
  const float hnew = (1.f - z) * n + z * h[j];

  mv[j] = hnew;
  __syncthreads();
  float s = 0.f, ss = 0.f;
  for (int c = 0; c < D_; ++c) { const float x = mv[c]; s += x; ss += x * x; }
  const float mean = s * (1.f / 64.f);
  const float rstd = rsqrtf(ss * (1.f / 64.f) - mean * mean + 1e-5f);
  const float mn = (hnew - mean) * rstd * gm[j] + bm[j];
  __syncthreads();
  mv[j] = mn;
  __syncthreads();

  float h0 = b1[j], h1 = b1[64 + j];
  const float* w1a = W1 + (size_t)j * D_;
  const float* w1b = W1 + (size_t)(64 + j) * D_;
  for (int c = 0; c < D_; ++c) { h0 += w1a[c] * mv[c]; h1 += w1b[c] * mv[c]; }
  hid[j] = fmaxf(h0, 0.f);
  hid[64 + j] = fmaxf(h1, 0.f);
  __syncthreads();

  float o = b2[j];
  const float* w2 = W2 + (size_t)j * H_;
  for (int k = 0; k < H_; ++k) o += w2[k] * hid[k];
  slots[(size_t)row * D_ + j] = hnew + o;
}

// ---------------------------------------------------------------------------
extern "C" void kernel_launch(void* const* d_in, const int* in_sizes, int n_in,
                              void* d_out, int out_size, void* d_ws, size_t ws_size,
                              hipStream_t stream) {
  const float* inputs      = (const float*)d_in[0];
  const float* noise       = (const float*)d_in[1];
  const float* gamma_in    = (const float*)d_in[2];
  const float* beta_in     = (const float*)d_in[3];
  const float* gamma_slots = (const float*)d_in[4];
  const float* beta_slots  = (const float*)d_in[5];
  const float* gamma_mlp   = (const float*)d_in[6];
  const float* beta_mlp    = (const float*)d_in[7];
  const float* slots_mu    = (const float*)d_in[8];
  const float* slots_lsig  = (const float*)d_in[9];
  const float* Wq          = (const float*)d_in[10];
  const float* Wk          = (const float*)d_in[11];
  const float* Wv          = (const float*)d_in[12];
  const float* W_ih        = (const float*)d_in[13];
  const float* W_hh        = (const float*)d_in[14];
  const float* b_ih        = (const float*)d_in[15];
  const float* b_hh        = (const float*)d_in[16];
  const float* W1          = (const float*)d_in[17];
  const float* b1          = (const float*)d_in[18];
  const float* W2          = (const float*)d_in[19];
  const float* b2          = (const float*)d_in[20];

  char* ws = (char*)d_ws;
  size_t off = 0;
  auto alloc = [&](size_t bytes) -> void* {
    void* p = (void*)(ws + off);
    off += (bytes + 255) & ~(size_t)255;
    return p;
  };
  _Float16* kf   = (_Float16*)alloc((size_t)B_ * N_ * D_ * 2);  // 67 MB
  _Float16* vT   = (_Float16*)alloc((size_t)B_ * D_ * N_ * 2);  // 67 MB
  _Float16* wk16 = (_Float16*)alloc((size_t)D_ * D_ * 2);
  _Float16* wv16 = (_Float16*)alloc((size_t)D_ * D_ * 2);
  _Float16* q16  = (_Float16*)alloc((size_t)B_ * 16 * D_ * 2);
  float*    updb = (float*)alloc((size_t)B_ * S_ * D_ * 4);

  float* slots = (float*)d_out;  // slots live in d_out, updated in place

  k_init_slots<<<(B_ * S_ * D_ + 255) / 256, 256, 0, stream>>>(
      noise, slots_mu, slots_lsig, slots, B_ * S_ * D_);
  k_w16<<<(D_ * D_ + 255) / 256, 256, 0, stream>>>(Wk, Wv, wk16, wv16);
  k_project<<<B_ * N_ / 128, 256, 0, stream>>>(inputs, gamma_in, beta_in,
                                               wk16, wv16, kf, vT);
  for (int it = 0; it < 3; ++it) {
    k_qproj<<<B_, 256, 0, stream>>>(slots, gamma_slots, beta_slots, Wq, q16);
    k_attention<<<B_, 256, 0, stream>>>(q16, kf, vT, updb);
    k_gru_mlp<<<B_ * S_, 64, 0, stream>>>(updb, slots, W_ih, W_hh, b_ih, b_hh,
                                          gamma_mlp, beta_mlp, W1, b1, W2, b2);
  }
}
